// StationFlowGAT_63015760166987
// MI455X (gfx1250) — matrix operand order repaired and checked by reading, not compiled, not run on hardware
//
#include <hip/hip_runtime.h>
#include <hip/hip_bf16.h>

typedef __attribute__((ext_vector_type(16))) __bf16 v16bf;
typedef __attribute__((ext_vector_type(8)))  float  v8f;

union FragBF {
  v16bf v;
  uint4 q[2];
};

#define NNODES 50000
#define NEDGES 800000

// ---------------------------------------------------------------------------
// WMMA GEMM:  C[M,NC] = act( A_rows @ W + bias ),  W given transposed Wt[NC][K] bf16.
// If iA != nullptr: row r = concat(F[iA[roff+r]][0..Khalf), F[iB[roff+r]][0..Khalf)),
// F row-stride = Khalf (A points at F), K = 2*Khalf.  K = NK*32.
// Block: (NC/16) waves; each wave owns one 16-col tile, loops 8 row tiles.
// REQUIREMENT (guaranteed by all launches below): M % 16 == 0.
// ---------------------------------------------------------------------------
template <int NK, int NC, int RELU, int OUTB>
__global__ void __launch_bounds__(256)
wmma_gemm_kernel(const __hip_bfloat16* __restrict__ A,
                 const __hip_bfloat16* __restrict__ Wt,
                 const float* __restrict__ bias,
                 float* __restrict__ Cf,
                 __hip_bfloat16* __restrict__ Cb,
                 int M,
                 const int* __restrict__ iA,
                 const int* __restrict__ iB,
                 int Khalf, int roff)
{
  constexpr int K = NK * 32;
  extern __shared__ char smem[];
  __hip_bfloat16* sW = (__hip_bfloat16*)smem;

  const int tid  = threadIdx.x;
  constexpr int NTHR  = 2 * NC;        // (NC/16) waves * 32
  constexpr int TOTAL = NC * K;
#pragma unroll
  for (int i = 0; i < TOTAL / 8; i += NTHR) {
    const int o = (i + tid) * 8;
    *(uint4*)(sW + o) = *(const uint4*)(Wt + o);
  }
  __syncthreads();

  const int wave  = tid >> 5;
  const int lane  = tid & 31;
  const int lrow  = lane & 15;   // A row-in-tile / B,C col
  const int lhalf = lane >> 4;   // lane half
  const int ncol0 = wave * 16;

  // B fragments: lane holds column ncol0+lrow; k = kt*32 + lhalf*16 + {0..15}
  FragBF bfrag[NK];
#pragma unroll
  for (int kt = 0; kt < NK; ++kt) {
    const __hip_bfloat16* bp = sW + (ncol0 + lrow) * K + kt * 32 + lhalf * 16;
    bfrag[kt].q[0] = *(const uint4*)(bp);
    bfrag[kt].q[1] = *(const uint4*)(bp + 8);
  }

  const float b = bias[ncol0 + lrow];

#pragma unroll 1
  for (int rt = 0; rt < 8; ++rt) {
    const int row0 = (blockIdx.x * 8 + rt) * 16;
    if (row0 >= M) return;                 // M % 16 == 0: full tiles only
    const int arow = row0 + lrow;

    const __hip_bfloat16* base0;
    const __hip_bfloat16* base1;
    if (iA) {
      base0 = A + (size_t)iA[roff + arow] * Khalf;
      base1 = A + (size_t)iB[roff + arow] * Khalf;
    } else {
      base0 = A + (size_t)arow * K;
      base1 = base0;
    }

    v8f acc = {};
#pragma unroll
    for (int kt = 0; kt < NK; ++kt) {
      // A fragment: k = kt*32 + lhalf*16 + {0..15}
      const int kbase = kt * 32 + lhalf * 16;
      const __hip_bfloat16* ap;
      if (iA) {
        ap = (kbase < Khalf) ? (base0 + kbase) : (base1 + (kbase - Khalf));
      } else {
        ap = base0 + kbase;
      }
      FragBF af;
      af.q[0] = *(const uint4*)(ap);
      af.q[1] = *(const uint4*)(ap + 8);
      acc = __builtin_amdgcn_wmma_f32_16x16x32_bf16(false, af.v, false, bfrag[kt].v,
                                                    (short)0, acc, false, false);
    }

    // C layout: VGPR r -> m = row0 + lhalf*8 + r, n = ncol0 + lrow
    size_t o = (size_t)(row0 + lhalf * 8) * NC + ncol0 + lrow;
#pragma unroll
    for (int r = 0; r < 8; ++r) {
      float v = acc[r] + b;
      if (RELU) v = fmaxf(v, 0.0f);
      if (OUTB) Cb[o] = __float2bfloat16(v);
      else      Cf[o] = v;
      o += NC;
    }
  }
}

// ---------------------------------------------------------------------------
// Small helper kernels
// ---------------------------------------------------------------------------
__global__ void f32_to_bf16_kernel(const float* __restrict__ in,
                                   __hip_bfloat16* __restrict__ out, int n)
{
  int i = blockIdx.x * blockDim.x + threadIdx.x;
  if (i < n) out[i] = __float2bfloat16(in[i]);
}

// W[K][Nc] f32 -> Wt[Nc][K] bf16 (transpose + convert)
__global__ void wt_bf16_kernel(const float* __restrict__ W,
                               __hip_bfloat16* __restrict__ T, int K, int Nc)
{
  int i = blockIdx.x * blockDim.x + threadIdx.x;
  if (i < K * Nc) {
    int n = i / K, k = i % K;
    T[i] = __float2bfloat16(W[(size_t)k * Nc + n]);
  }
}

__global__ void init_ms_kernel(unsigned* __restrict__ m_u, float* __restrict__ s, int n)
{
  int i = blockIdx.x * blockDim.x + threadIdx.x;
  if (i < n) { m_u[i] = 0u; s[i] = 0.0f; }
}

__global__ void init_out_kernel(float* __restrict__ out, const float* __restrict__ bo,
                                int n, int HC)
{
  int i = blockIdx.x * blockDim.x + threadIdx.x;
  if (i < n) out[i] = bo[i % HC];
}

__global__ void attn_logits_kernel(const float* __restrict__ xl, const float* __restrict__ xr,
                                   const float* __restrict__ att,
                                   const int* __restrict__ src, const int* __restrict__ dst,
                                   int E, int Eall, int C,
                                   float* __restrict__ logit, unsigned* __restrict__ m_u)
{
  int e = blockIdx.x * blockDim.x + threadIdx.x;
  if (e >= Eall) return;
  const int sn = e < E ? src[e] : e - E;
  const int dn = e < E ? dst[e] : e - E;
  const float* pl = xl + (size_t)sn * 2 * C;
  const float* pr = xr + (size_t)dn * 2 * C;
#pragma unroll
  for (int h = 0; h < 2; ++h) {
    float acc = 0.0f;
    for (int c = 0; c < C; ++c) {
      float t = pl[h * C + c] + pr[h * C + c];
      t = t > 0.0f ? t : 0.2f * t;   // leaky_relu, slope 0.2
      acc += att[h * C + c] * t;
    }
    logit[(size_t)e * 2 + h] = acc;
    // order-preserving float->uint map for atomic max
    unsigned bits = __float_as_uint(acc);
    unsigned key  = bits ^ (unsigned)(((int)bits >> 31) | 0x80000000);
    atomicMax(&m_u[dn * 2 + h], key);
  }
}

__global__ void attn_exp_kernel(float* __restrict__ logit, const unsigned* __restrict__ m_u,
                                const int* __restrict__ dst, int E, int Eall,
                                float* __restrict__ s)
{
  int e = blockIdx.x * blockDim.x + threadIdx.x;
  if (e >= Eall) return;
  const int dn = e < E ? dst[e] : e - E;
#pragma unroll
  for (int h = 0; h < 2; ++h) {
    unsigned key = m_u[dn * 2 + h];
    unsigned mb  = (key & 0x80000000u) ? (key ^ 0x80000000u) : ~key;
    float m = __uint_as_float(mb);
    float p = __expf(logit[(size_t)e * 2 + h] - m);
    logit[(size_t)e * 2 + h] = p;
    atomicAdd(&s[dn * 2 + h], p);
  }
}

__global__ void attn_aggr_kernel(const float* __restrict__ p, const float* __restrict__ s,
                                 const float* __restrict__ xl,
                                 const int* __restrict__ src, const int* __restrict__ dst,
                                 int E, int Eall, int C, float* __restrict__ out)
{
  long long t = (long long)blockIdx.x * blockDim.x + threadIdx.x;
  const int HC = 2 * C;
  if (t >= (long long)Eall * HC) return;
  const int e = (int)(t / HC);
  const int j = (int)(t % HC);
  const int h = j / C;
  const int sn = e < E ? src[e] : e - E;
  const int dn = e < E ? dst[e] : e - E;
  const float alpha = p[(size_t)e * 2 + h] / s[dn * 2 + h];
  atomicAdd(&out[(size_t)dn * HC + j], alpha * xl[(size_t)sn * HC + j]);
}

__global__ void relu_bf16_kernel(const float* __restrict__ in,
                                 __hip_bfloat16* __restrict__ out, int n)
{
  int i = blockIdx.x * blockDim.x + threadIdx.x;
  if (i < n) out[i] = __float2bfloat16(fmaxf(in[i], 0.0f));
}

// Nc==1 head layer: out[i] = relu( dot(in[i], w) + b )
template <int INBF>
__global__ void dot_head_kernel(const float* __restrict__ inF,
                                const __hip_bfloat16* __restrict__ inB,
                                const float* __restrict__ w,
                                const float* __restrict__ bptr,
                                float* __restrict__ out, int M, int K)
{
  int i = blockIdx.x * blockDim.x + threadIdx.x;
  if (i >= M) return;
  float acc = bptr[0];
  for (int k = 0; k < K; ++k) {
    float a = INBF ? __bfloat162float(inB[(size_t)i * K + k]) : inF[(size_t)i * K + k];
    acc += a * w[k];
  }
  out[i] = fmaxf(acc, 0.0f);
}

// ---------------------------------------------------------------------------
// GEMM dispatch (all used template combos)
// ---------------------------------------------------------------------------
static void launch_gemm(int NK, int Nc, bool relu, bool outb,
                        const __hip_bfloat16* A, const __hip_bfloat16* Wt, const float* bias,
                        float* Cf, __hip_bfloat16* Cb, int M,
                        const int* iA, const int* iB, int Khalf, int roff,
                        hipStream_t stream)
{
  dim3 blk(32 * (Nc / 16));
  dim3 grd((M + 127) / 128);
  size_t sh = (size_t)Nc * (NK * 32) * 2;
#define GEMM_ARGS A, Wt, bias, Cf, Cb, M, iA, iB, Khalf, roff
  if      (NK == 4 && Nc == 128 && !relu && !outb)
    wmma_gemm_kernel<4,128,0,0><<<grd, blk, sh, stream>>>(GEMM_ARGS);
  else if (NK == 4 && Nc == 64 && !relu && !outb)
    wmma_gemm_kernel<4, 64,0,0><<<grd, blk, sh, stream>>>(GEMM_ARGS);
  else if (NK == 2 && Nc == 64 && !relu && !outb)
    wmma_gemm_kernel<2, 64,0,0><<<grd, blk, sh, stream>>>(GEMM_ARGS);
  else if (NK == 2 && Nc == 32 && relu && !outb)
    wmma_gemm_kernel<2, 32,1,0><<<grd, blk, sh, stream>>>(GEMM_ARGS);
  else if (NK == 4 && Nc == 64 && relu && outb)
    wmma_gemm_kernel<4, 64,1,1><<<grd, blk, sh, stream>>>(GEMM_ARGS);
  else if (NK == 2 && Nc == 32 && relu && outb)
    wmma_gemm_kernel<2, 32,1,1><<<grd, blk, sh, stream>>>(GEMM_ARGS);
  else if (NK == 1 && Nc == 32 && relu && outb)
    wmma_gemm_kernel<1, 32,1,1><<<grd, blk, sh, stream>>>(GEMM_ARGS);
#undef GEMM_ARGS
}

// ---------------------------------------------------------------------------
extern "C" void kernel_launch(void* const* d_in, const int* in_sizes, int n_in,
                              void* d_out, int out_size, void* d_ws, size_t ws_size,
                              hipStream_t stream)
{
  const int N = NNODES, E = NEDGES, Eall = E + N;
  (void)in_sizes; (void)n_in; (void)out_size; (void)ws_size;

  const float* x    = (const float*)d_in[0];
  const int*   ei   = (const int*)d_in[1];
  const int*   srcI = ei;
  const int*   dstI = ei + E;

  const float* Wl[3]  = {(const float*)d_in[2],  (const float*)d_in[8],  (const float*)d_in[14]};
  const float* bl[3]  = {(const float*)d_in[3],  (const float*)d_in[9],  (const float*)d_in[15]};
  const float* Wr[3]  = {(const float*)d_in[4],  (const float*)d_in[10], (const float*)d_in[16]};
  const float* br[3]  = {(const float*)d_in[5],  (const float*)d_in[11], (const float*)d_in[17]};
  const float* att[3] = {(const float*)d_in[6],  (const float*)d_in[12], (const float*)d_in[18]};
  const float* bo[3]  = {(const float*)d_in[7],  (const float*)d_in[13], (const float*)d_in[19]};
  const float* Wn1 = (const float*)d_in[20];
  const float* bn1 = (const float*)d_in[21];
  const float* Wn2 = (const float*)d_in[22];
  const float* bn2 = (const float*)d_in[23];
  const float* We[4] = {(const float*)d_in[24], (const float*)d_in[26],
                        (const float*)d_in[28], (const float*)d_in[30]};
  const float* be[4] = {(const float*)d_in[25], (const float*)d_in[27],
                        (const float*)d_in[29], (const float*)d_in[31]};

  // ---- workspace carve -----------------------------------------------------
  char* wp = (char*)d_ws;
  auto carve = [&](size_t bytes) -> void* {
    void* p = (void*)wp;
    wp += (bytes + 255) & ~(size_t)255;
    return p;
  };
  __hip_bfloat16* hb   = (__hip_bfloat16*)carve((size_t)N * 128 * 2); // bf16 features (reused each layer)
  float* xl            = (float*)carve((size_t)N * 128 * 4);
  float* xr            = (float*)carve((size_t)N * 128 * 4);
  float* outa          = (float*)carve((size_t)N * 128 * 4);
  float* logit         = (float*)carve((size_t)Eall * 2 * 4);
  unsigned* m_u        = (unsigned*)carve((size_t)N * 2 * 4);
  float* ssum          = (float*)carve((size_t)N * 2 * 4);
  __hip_bfloat16* tWl[3], *tWr[3];
  tWl[0] = (__hip_bfloat16*)carve(128 * 128 * 2);
  tWr[0] = (__hip_bfloat16*)carve(128 * 128 * 2);
  tWl[1] = (__hip_bfloat16*)carve(64 * 128 * 2);
  tWr[1] = (__hip_bfloat16*)carve(64 * 128 * 2);
  tWl[2] = (__hip_bfloat16*)carve(64 * 64 * 2);
  tWr[2] = (__hip_bfloat16*)carve(64 * 64 * 2);
  __hip_bfloat16* tWn1 = (__hip_bfloat16*)carve(32 * 64 * 2);
  __hip_bfloat16* tWe1 = (__hip_bfloat16*)carve(64 * 128 * 2);
  __hip_bfloat16* tWe2 = (__hip_bfloat16*)carve(32 * 64 * 2);
  __hip_bfloat16* tWe3 = (__hip_bfloat16*)carve(32 * 32 * 2);

  auto cvtW = [&](const float* W, __hip_bfloat16* T, int K, int Nc) {
    int n = K * Nc;
    wt_bf16_kernel<<<(n + 255) / 256, 256, 0, stream>>>(W, T, K, Nc);
  };

  // ---- weight conversion (f32 -> transposed bf16) --------------------------
  cvtW(Wl[0], tWl[0], 128, 128); cvtW(Wr[0], tWr[0], 128, 128);
  cvtW(Wl[1], tWl[1], 128, 64);  cvtW(Wr[1], tWr[1], 128, 64);
  cvtW(Wl[2], tWl[2], 64, 64);   cvtW(Wr[2], tWr[2], 64, 64);
  cvtW(Wn1, tWn1, 64, 32);
  cvtW(We[0], tWe1, 128, 64);
  cvtW(We[1], tWe2, 64, 32);
  cvtW(We[2], tWe3, 32, 32);

  // ---- input features -> bf16 ---------------------------------------------
  f32_to_bf16_kernel<<<(N * 128 + 255) / 256, 256, 0, stream>>>(x, hb, N * 128);

  // ---- 3 GATv2 layers ------------------------------------------------------
  const int dinL[3] = {128, 128, 64};
  const int CcL[3]  = {64, 32, 32};
  for (int l = 0; l < 3; ++l) {
    const int C  = CcL[l];
    const int HC = 2 * C;
    const int NK = dinL[l] / 32;
    launch_gemm(NK, HC, false, false, hb, tWl[l], bl[l], xl, nullptr, N,
                nullptr, nullptr, 0, 0, stream);
    launch_gemm(NK, HC, false, false, hb, tWr[l], br[l], xr, nullptr, N,
                nullptr, nullptr, 0, 0, stream);
    init_ms_kernel<<<(N * 2 + 255) / 256, 256, 0, stream>>>(m_u, ssum, N * 2);
    init_out_kernel<<<(N * HC + 255) / 256, 256, 0, stream>>>(outa, bo[l], N * HC, HC);
    attn_logits_kernel<<<(Eall + 255) / 256, 256, 0, stream>>>(xl, xr, att[l], srcI, dstI,
                                                               E, Eall, C, logit, m_u);
    attn_exp_kernel<<<(Eall + 255) / 256, 256, 0, stream>>>(logit, m_u, dstI, E, Eall, ssum);
    long long tot = (long long)Eall * HC;
    attn_aggr_kernel<<<(unsigned)((tot + 255) / 256), 256, 0, stream>>>(logit, ssum, xl, srcI,
                                                                        dstI, E, Eall, C, outa);
    relu_bf16_kernel<<<(N * HC + 255) / 256, 256, 0, stream>>>(outa, hb, N * HC);
  }

  // ---- node head: relu(relu(h @ Wn1 + bn1) @ Wn2 + bn2) -------------------
  float* hn1 = outa; // alias: outa free after layer 3
  launch_gemm(2, 32, true, false, hb, tWn1, bn1, hn1, nullptr, N,
              nullptr, nullptr, 0, 0, stream);
  dot_head_kernel<0><<<(N + 255) / 256, 256, 0, stream>>>(hn1, nullptr, Wn2, bn2,
                                                          (float*)d_out, N, 32);

  // ---- edge head (chunked, buffers aliased into freed GAT scratch) --------
  const int CH = 131072;                 // multiple of 16; E % 16 == 0 so tail is too
  __hip_bfloat16* he1 = (__hip_bfloat16*)xl;                          // CH*64 bf16 (16.8 MB)
  __hip_bfloat16* he2 = (__hip_bfloat16*)xr;                          // CH*32 bf16
  __hip_bfloat16* he3 = (__hip_bfloat16*)((char*)xr + (size_t)CH * 32 * 2);
  float* outE = (float*)d_out + N;
  for (int base = 0; base < E; base += CH) {
    const int mc = (E - base) < CH ? (E - base) : CH;
    // layer1: gathered concat(h[src], h[dst]) : K=128 (Khalf=64), Nc=64
    launch_gemm(4, 64, true, true, hb, tWe1, be[0], nullptr, he1, mc,
                srcI, dstI, 64, base, stream);
    launch_gemm(2, 32, true, true, he1, tWe2, be[1], nullptr, he2, mc,
                nullptr, nullptr, 0, 0, stream);
    launch_gemm(1, 32, true, true, he2, tWe3, be[2], nullptr, he3, mc,
                nullptr, nullptr, 0, 0, stream);
    dot_head_kernel<1><<<(mc + 255) / 256, 256, 0, stream>>>(nullptr, he3, We[3], be[3],
                                                             outE + base, mc, 32);
  }
}
